// TotalNet_29472065585224
// MI455X (gfx1250) — matrix operand order repaired and checked by reading, not compiled, not run on hardware
//
#include <hip/hip_runtime.h>
#include <cstdint>
#include <cstddef>

typedef _Float16 half_t;
typedef __attribute__((ext_vector_type(16))) _Float16 v16h;
typedef __attribute__((ext_vector_type(8)))  _Float16 v8h;
typedef __attribute__((ext_vector_type(8)))  float    v8f;

#define BM   64      // batch rows per workgroup (2 waves x 32 rows)
#define LDH  136     // f16 row stride in LDS activation tiles (16B aligned)
#define LDF  33      // f32 row stride for residual buffer

// ---------------- const-block (f32) layout in workspace ----------------
static constexpr int CB_RES     = 0;     // 80 floats of conv params (res)
static constexpr int CB_RES2    = 80;    // 80 floats (res2)
static constexpr int CB_R1_HB1  = 160;   // 64
static constexpr int CB_R1_HB2  = 224;   // 32
static constexpr int CB_R2_HB1  = 256;   // 64
static constexpr int CB_R2_HB2  = 320;   // 32
static constexpr int CB_META    = 352;   // 584
static constexpr int CB_HB1     = 936;   // 128
static constexpr int CB_HB2     = 1064;  // 128
static constexpr int CB_HB3     = 1192;  // 128
static constexpr int CB_HB4     = 1320;  // 32  (total 1352 floats)

// ---------------- packed f16 WMMA B-fragment offsets (in halves) -------
static constexpr int WF_R1W1 = 0;       // K=96  N=64  -> 3*4*512
static constexpr int WF_R1W2 = 6144;    // K=64  N=32  -> 2*2*512
static constexpr int WF_R2W1 = 8192;
static constexpr int WF_R2W2 = 14336;
static constexpr int WF_HW1  = 16384;   // K=39 pad 64, N=128 -> 2*8*512
static constexpr int WF_HW2  = 24576;   // 128x128 -> 4*8*512
static constexpr int WF_HW3  = 40960;
static constexpr int WF_HW4  = 57344;   // 128x32 -> 4*2*512 (total 61440 halves)

// ---------------- dilated conv specs -----------------------------------
__device__ constexpr int N1_DIL[7] = {1,1,2,4,8,16,32};
__device__ constexpr int N1_PAD[7] = {1,0,1,2,4,8,16};
__device__ constexpr int N2_DIL[7] = {1,1,2,4,8,16,32};
__device__ constexpr int N2_PAD[7] = {1,2,3,6,12,24,48};
__device__ constexpr int N3_DIL[6] = {1,2,4,8,16,32};
__device__ constexpr int N3_PAD[6] = {1,2,4,8,16,32};

// gfx1250 has a hardware V_TANH_F32 (TRANS32); use it when the builtin
// exists, otherwise fall back to exp + v_rcp (no IEEE division sequence).
__device__ __forceinline__ float fast_tanh(float x) {
#if __has_builtin(__builtin_amdgcn_tanhf)
  return __builtin_amdgcn_tanhf(x);
#else
  float e = __expf(2.0f * x);
#if __has_builtin(__builtin_amdgcn_rcpf)
  float r = __builtin_amdgcn_rcpf(e + 1.0f);
#else
  float r = 1.0f / (e + 1.0f);
#endif
  return 1.0f - 2.0f * r;
#endif
}

// ---- conv chain: runs one net on a 32-sample sequence (LDS scratch) ----
template <int K, int NL>
__device__ __forceinline__ void chain_run(const float* __restrict__ w,
                                          const float* __restrict__ b,
                                          const int (&dil)[NL], const int (&pad)[NL],
                                          const float* __restrict__ in,
                                          float* s, float* d, half_t* outH)
{
  int len = 32;
#pragma unroll
  for (int i = 0; i < 32; ++i) s[i] = in[i];
#pragma unroll
  for (int l = 0; l < NL; ++l) {
    const int di = dil[l], pp = pad[l];
    const int olen = len + 2 * pp - di * (K - 1);
    const float bv = b[l];
    float wl[K];
#pragma unroll
    for (int j = 0; j < K; ++j) wl[j] = w[l * K + j];
    for (int i = 0; i < olen; ++i) {
      float acc = bv;
#pragma unroll
      for (int j = 0; j < K; ++j) {
        int idx = i - pp + j * di;
        float v = (idx >= 0 && idx < len) ? s[idx] : 0.0f;
        acc = fmaf(wl[j], v, acc);
      }
      if (l < 3)            acc = fmaxf(acc, 0.0f);   // relu x3
      else if (l != NL - 1) acc = fast_tanh(acc);     // tanh middle, none last
      d[i] = acc;
    }
    len = olen;
    float* t = s; s = d; d = t;
  }
#pragma unroll
  for (int i = 0; i < 32; ++i) outH[i] = (half_t)s[i];
}

__device__ __forceinline__ void run_dilated(const float* __restrict__ cp,
                                            const float* __restrict__ in,
                                            float* cs, float* cd, half_t* rowH)
{
  chain_run<2, 7>(cp + 0,  cp + 14, N1_DIL, N1_PAD, in, cs, cd, rowH + 0);
  chain_run<4, 7>(cp + 21, cp + 49, N2_DIL, N2_PAD, in, cs, cd, rowH + 32);
  chain_run<3, 6>(cp + 56, cp + 74, N3_DIL, N3_PAD, in, cs, cd, rowH + 64);
}

// ---- meta net: (8x5) -> 7 outputs, pure VALU --------------------------
__device__ __forceinline__ void meta_row(const float* __restrict__ mp,
                                         const float* __restrict__ mrow, float mo[7])
{
  const float* w1 = mp;          // [5][32]
  const float* b1 = mp + 160;
  const float* w2 = mp + 192;    // [32][7]
  const float* b2 = mp + 416;
  const float* w3 = mp + 423;    // [8][16]
  const float* b3 = mp + 551;
  const float* w4 = mp + 567;    // [16]
  const float  b4 = mp[583];
  float mm[40];
#pragma unroll
  for (int i = 0; i < 40; ++i) mm[i] = mrow[i];
  float h2[8][7];
#pragma unroll
  for (int i = 0; i < 8; ++i) {
    float acc[7];
#pragma unroll
    for (int j = 0; j < 7; ++j) acc[j] = b2[j];
    for (int c = 0; c < 32; ++c) {
      float t = b1[c];
#pragma unroll
      for (int f = 0; f < 5; ++f) t = fmaf(mm[i * 5 + f], w1[f * 32 + c], t);
      t = fast_tanh(t);
#pragma unroll
      for (int j = 0; j < 7; ++j) acc[j] = fmaf(t, w2[c * 7 + j], acc[j]);
    }
#pragma unroll
    for (int j = 0; j < 7; ++j) h2[i][j] = acc[j];
  }
#pragma unroll
  for (int j = 0; j < 7; ++j) {          // after swapaxes: row j of (7x8)
    float o = b4;
    for (int c = 0; c < 16; ++c) {
      float g = b3[c];
#pragma unroll
      for (int i = 0; i < 8; ++i) g = fmaf(h2[i][j], w3[i * 16 + c], g);
      o = fmaf(fast_tanh(g), w4[c], o);
    }
    mo[j] = o;
  }
}

// ---- A-fragment load from row-major f16 LDS tile (ISA 16x32 layout) ---
__device__ __forceinline__ v16h lds_a_frag(const half_t* A, int rowBase, int kBase, int lane)
{
  const int m  = lane & 15;
  const int hh = lane >> 4;
  const half_t* p = A + (rowBase + m) * LDH + kBase + hh * 8;
  v8h lo = *(const v8h*)(p);        // K = kBase + hh*8 .. +7
  v8h hi = *(const v8h*)(p + 16);   // K = kBase + 16 + hh*8 .. +7
  v16h a;
#pragma unroll
  for (int i = 0; i < 8; ++i) { a[i] = lo[i]; a[i + 8] = hi[i]; }
  return a;
}

// ---- per-wave GEMM over its 32-row strip ------------------------------
// MODE 0: tanh -> f16 LDS (stride LDH);  MODE 1: raw -> f32 LDS (stride LDF)
// MODE 2: 1.0f + v -> f32 global (stride 32)
template <int KT, int NTI, int MODE>
__device__ __forceinline__ void wave_gemm(const half_t* __restrict__ A,
                                          const half_t* __restrict__ WF,
                                          const float* __restrict__ bias,
                                          half_t* outH, float* outF, float* outG,
                                          int rowBase, int lane)
{
  const int nloc = lane & 15;
  const int hh   = lane >> 4;
#pragma unroll
  for (int mt = 0; mt < 2; ++mt) {
    const int rb = rowBase + mt * 16;
#pragma unroll
    for (int nt = 0; nt < NTI; ++nt) {
      const float bv = bias[nt * 16 + nloc];   // bias depends only on N
      v8f c;
#pragma unroll
      for (int r = 0; r < 8; ++r) c[r] = bv;
#pragma unroll
      for (int kt = 0; kt < KT; ++kt) {
        v16h a = lds_a_frag(A, rb, kt * 32, lane);
        v16h b = *(const v16h*)(WF + ((size_t)(kt * NTI + nt) * 32 + lane) * 16);
        c = __builtin_amdgcn_wmma_f32_16x16x32_f16(false, a, false, b,
                                                   (short)0, c, false, false);
      }
      const int n = nt * 16 + nloc;
#pragma unroll
      for (int r = 0; r < 8; ++r) {
        const int row = rb + hh * 8 + r;       // C/D layout: M = r + 8*(lane/16)
        const float v = c[r];
        if (MODE == 0)      outH[row * LDH + n] = (half_t)fast_tanh(v);
        else if (MODE == 1) outF[row * LDF + n] = v;
        else                outG[row * 32  + n] = 1.0f + v;
      }
    }
  }
}

// ======================= fused main kernel =============================
__global__ __launch_bounds__(BM)
void totalnet_main(const float* __restrict__ x, const float* __restrict__ met,
                   const float* __restrict__ cb, const half_t* __restrict__ wf,
                   float* __restrict__ out)
{
  __shared__ __align__(16) half_t bufA[BM * LDH];
  __shared__ __align__(16) half_t bufB[BM * LDH];
  __shared__ float bufF[BM * LDF];
  __shared__ float chainS[BM * 70];

  const int    tid     = threadIdx.x;
  const int    lane    = tid & 31;
  const int    rowBase = (tid >> 5) * 32;
  const size_t gRow    = (size_t)blockIdx.x * BM + tid;

  float* cs = &chainS[tid * 70];
  float* cd = cs + 35;

  float xa[32];
#pragma unroll
  for (int c = 0; c < 32; ++c) xa[c] = x[gRow * 32 + c];

  // ---------- residual dilated block 1 ----------
  run_dilated(cb + CB_RES, xa, cs, cd, &bufA[tid * LDH]);      // h96 (f16)
  __syncthreads();
  wave_gemm<3, 4, 0>(bufA, wf + WF_R1W1, cb + CB_R1_HB1, bufB, nullptr, nullptr, rowBase, lane);
  __syncthreads();
  wave_gemm<2, 2, 1>(bufB, wf + WF_R1W2, cb + CB_R1_HB2, nullptr, bufF, nullptr, rowBase, lane);
  __syncthreads();

  float oa[32];
#pragma unroll
  for (int c = 0; c < 32; ++c) oa[c] = bufF[tid * LDF + c] + xa[c];

  // ---------- residual dilated block 2 ----------
  run_dilated(cb + CB_RES2, oa, cs, cd, &bufA[tid * LDH]);
  __syncthreads();
  wave_gemm<3, 4, 0>(bufA, wf + WF_R2W1, cb + CB_R2_HB1, bufB, nullptr, nullptr, rowBase, lane);
  __syncthreads();
  wave_gemm<2, 2, 1>(bufB, wf + WF_R2W2, cb + CB_R2_HB2, nullptr, bufF, nullptr, rowBase, lane);
  __syncthreads();

  // ---------- concat [o2(32) | meta(7) | zero-pad to 64] ----------
  half_t* rowp = &bufA[tid * LDH];
#pragma unroll
  for (int c = 0; c < 32; ++c) rowp[c] = (half_t)(bufF[tid * LDF + c] + oa[c]);
  float mo[7];
  meta_row(cb + CB_META, met + gRow * 40, mo);
#pragma unroll
  for (int j = 0; j < 7; ++j) rowp[32 + j] = (half_t)mo[j];
#pragma unroll
  for (int c = 39; c < 64; ++c) rowp[c] = (half_t)0.0f;   // keep padded K clean
  __syncthreads();

  // ---------- head MLP: 39(p64)->128->128->128->32 ----------
  wave_gemm<2, 8, 0>(bufA, wf + WF_HW1, cb + CB_HB1, bufB, nullptr, nullptr, rowBase, lane);
  __syncthreads();
  wave_gemm<4, 8, 0>(bufB, wf + WF_HW2, cb + CB_HB2, bufA, nullptr, nullptr, rowBase, lane);
  __syncthreads();
  wave_gemm<4, 8, 0>(bufA, wf + WF_HW3, cb + CB_HB3, bufB, nullptr, nullptr, rowBase, lane);
  __syncthreads();
  wave_gemm<4, 2, 2>(bufB, wf + WF_HW4, cb + CB_HB4, nullptr, nullptr,
                     out + (size_t)blockIdx.x * BM * 32, rowBase, lane);
}

// ======================= weight prep kernel ============================
struct ParamPtrs { const float* p[107]; };

__device__ __forceinline__ void cpy(float* d, const float* s, int n, int tid, int nth) {
  for (int i = tid; i < n; i += nth) d[i] = s[i];
}

// Pack W[K][N] (f32 row-major) into exact WMMA B-fragment order:
// frag (kt,nt), lane L, elem e  <->  K = kt*32 + (e<8 ? (L/16)*8 + e : 16 + (L/16)*8 + e-8), N = nt*16 + (L&15)
__device__ void pack_wmma_b(half_t* dst, const float* W, int Kreal, int Kpad, int N,
                            int tid, int nth)
{
  const int KT = Kpad >> 5, NTt = N >> 4;
  const int tot = KT * NTt * 512;
  for (int o = tid; o < tot; o += nth) {
    const int e  = o & 15;
    const int L  = (o >> 4) & 31;
    const int nt = (o >> 9) % NTt;
    const int kt = o / (512 * NTt);
    const int K  = kt * 32 + ((e < 8) ? (L >> 4) * 8 + e : 16 + (L >> 4) * 8 + (e - 8));
    const int n  = nt * 16 + (L & 15);
    const float v = (K < Kreal) ? W[K * N + n] : 0.0f;
    dst[o] = (half_t)v;
  }
}

__global__ void totalnet_prep(ParamPtrs pp, float* cb, half_t* wf)
{
  const int tid = threadIdx.x, nth = blockDim.x;
  // conv params for res (base 3) and res2 (base 47)
  for (int net = 0; net < 2; ++net) {
    const int base = 3 + net * 44;
    float* D = cb + net * 80;
    for (int i = 0; i < 7; ++i) { cpy(D + 2 * i,      pp.p[base + 2 * i],      2, tid, nth);
                                  cpy(D + 14 + i,     pp.p[base + 1 + 2 * i],  1, tid, nth); }
    for (int i = 0; i < 7; ++i) { cpy(D + 21 + 4 * i, pp.p[base + 14 + 2 * i], 4, tid, nth);
                                  cpy(D + 49 + i,     pp.p[base + 15 + 2 * i], 1, tid, nth); }
    for (int i = 0; i < 6; ++i) { cpy(D + 56 + 3 * i, pp.p[base + 28 + 2 * i], 3, tid, nth);
                                  cpy(D + 74 + i,     pp.p[base + 29 + 2 * i], 1, tid, nth); }
  }
  cpy(cb + CB_R1_HB1, pp.p[44], 64, tid, nth);
  cpy(cb + CB_R1_HB2, pp.p[46], 32, tid, nth);
  cpy(cb + CB_R2_HB1, pp.p[88], 64, tid, nth);
  cpy(cb + CB_R2_HB2, pp.p[90], 32, tid, nth);
  cpy(cb + CB_META + 0,   pp.p[91], 160, tid, nth);
  cpy(cb + CB_META + 160, pp.p[92], 32,  tid, nth);
  cpy(cb + CB_META + 192, pp.p[93], 224, tid, nth);
  cpy(cb + CB_META + 416, pp.p[94], 7,   tid, nth);
  cpy(cb + CB_META + 423, pp.p[95], 128, tid, nth);
  cpy(cb + CB_META + 551, pp.p[96], 16,  tid, nth);
  cpy(cb + CB_META + 567, pp.p[97], 16,  tid, nth);
  cpy(cb + CB_META + 583, pp.p[98], 1,   tid, nth);
  cpy(cb + CB_HB1, pp.p[100], 128, tid, nth);
  cpy(cb + CB_HB2, pp.p[102], 128, tid, nth);
  cpy(cb + CB_HB3, pp.p[104], 128, tid, nth);
  cpy(cb + CB_HB4, pp.p[106], 32,  tid, nth);

  pack_wmma_b(wf + WF_R1W1, pp.p[43],  96,  96,  64, tid, nth);
  pack_wmma_b(wf + WF_R1W2, pp.p[45],  64,  64,  32, tid, nth);
  pack_wmma_b(wf + WF_R2W1, pp.p[87],  96,  96,  64, tid, nth);
  pack_wmma_b(wf + WF_R2W2, pp.p[89],  64,  64,  32, tid, nth);
  pack_wmma_b(wf + WF_HW1,  pp.p[99],  39,  64, 128, tid, nth);
  pack_wmma_b(wf + WF_HW2,  pp.p[101], 128, 128, 128, tid, nth);
  pack_wmma_b(wf + WF_HW3,  pp.p[103], 128, 128, 128, tid, nth);
  pack_wmma_b(wf + WF_HW4,  pp.p[105], 128, 128,  32, tid, nth);
}

// ======================= launch ========================================
extern "C" void kernel_launch(void* const* d_in, const int* in_sizes, int n_in,
                              void* d_out, int out_size, void* d_ws, size_t ws_size,
                              hipStream_t stream)
{
  (void)out_size; (void)ws_size;
  ParamPtrs pp;
  for (int i = 0; i < 107; ++i) pp.p[i] = (i < n_in) ? (const float*)d_in[i] : nullptr;

  float*  cb = (float*)d_ws;                        // 1352 f32 const block
  half_t* wf = (half_t*)((char*)d_ws + 8192);       // 61440 f16 weight frags

  totalnet_prep<<<1, 256, 0, stream>>>(pp, cb, wf);

  const int Btot = in_sizes[0] / 32;
  const int nblk = Btot / BM;
  totalnet_main<<<nblk, BM, 0, stream>>>((const float*)d_in[0], (const float*)d_in[1],
                                         cb, wf, (float*)d_out);
}
